// multihead_attention_5746666242378
// MI455X (gfx1250) — matrix operand order repaired and checked by reading, not compiled, not run on hardware
//
#include <hip/hip_runtime.h>

// ---------------------------------------------------------------------------
// Multi-head attention for MI455X (gfx1250), bf16 WMMA pipeline.
// S=2048, B=4, F=1024, P=1024, H=16, D=64
// ---------------------------------------------------------------------------

#define S_LEN 2048
#define BATCH 4
#define FDIM  1024
#define PDIM  1024
#define NHEAD 16
#define HDIM  64
#define MROWS (S_LEN * BATCH)   // 8192 rows for all projection GEMMs

// LDS tile strides (elements), padded to spread banks, 16B-aligned rows
#define KSTRIDE 72   // K tile rows of 64 bf16 (+8 pad)
#define VSTRIDE 40   // V tile rows of 32 bf16 (+8 pad)
#define PSTRIDE 40   // P staging rows of 32 bf16 (+8 pad)

#if defined(__AMDGCN__) && \
    __has_builtin(__builtin_amdgcn_global_load_async_to_lds_b128) && \
    __has_builtin(__builtin_amdgcn_s_wait_asynccnt)
#define USE_ASYNC_LDS 1
#else
#define USE_ASYNC_LDS 0
#endif

typedef __attribute__((ext_vector_type(16))) __bf16 v16bf;
typedef __attribute__((ext_vector_type(8)))  __bf16 v8bf;
typedef __attribute__((ext_vector_type(8)))  float  v8f;
typedef __attribute__((ext_vector_type(4)))  int    v4i;

__device__ __forceinline__ unsigned short f2bf(float x) {
  unsigned int u = __float_as_uint(x);
  u += 0x7FFFu + ((u >> 16) & 1u);   // round-to-nearest-even
  return (unsigned short)(u >> 16);
}

// A-matrix 16x32 bf16 fragment (ISA 7.12.2): lo-half lane holds row m,
// K = {k..k+7, k+16..k+23}; hi half +8. p = row_base + kbase + half*8.
__device__ __forceinline__ v16bf load_afrag(const unsigned short* p) {
  v8bf a = *(const v8bf*)(p);
  v8bf b = *(const v8bf*)(p + 16);
  v16bf r;
#pragma unroll
  for (int i = 0; i < 8; ++i) { r[i] = a[i]; r[i + 8] = b[i]; }
  return r;
}

// B-matrix 32x16 bf16 fragment: lane = column n; same K order as A:
// K = {k..k+7, k+16..k+23}; hi half +8. p = col_row_base + kbase + half*8.
__device__ __forceinline__ v16bf load_bfrag(const unsigned short* p) {
  v8bf a = *(const v8bf*)(p);
  v8bf b = *(const v8bf*)(p + 16);
  v16bf r;
#pragma unroll
  for (int i = 0; i < 8; ++i) { r[i] = a[i]; r[i + 8] = b[i]; }
  return r;
}

__device__ __forceinline__ v8f wmma_bf16(v16bf a, v16bf b, v8f c) {
  return __builtin_amdgcn_wmma_f32_16x16x32_bf16(false, a, false, b,
                                                 (short)0, c, false, false);
}

__device__ __forceinline__ float hmax16(float v) {
  v = fmaxf(v, __shfl_xor(v, 1, 32));
  v = fmaxf(v, __shfl_xor(v, 2, 32));
  v = fmaxf(v, __shfl_xor(v, 4, 32));
  v = fmaxf(v, __shfl_xor(v, 8, 32));
  return v;
}
__device__ __forceinline__ float hsum16(float v) {
  v += __shfl_xor(v, 1, 32);
  v += __shfl_xor(v, 2, 32);
  v += __shfl_xor(v, 4, 32);
  v += __shfl_xor(v, 8, 32);
  return v;
}

// 16-byte global -> LDS copy; async (no VGPR round trip) when available.
__device__ __forceinline__ void copy16_g2l(const unsigned short* g,
                                           unsigned short* l) {
#if USE_ASYNC_LDS
  __builtin_amdgcn_global_load_async_to_lds_b128(
      (__attribute__((address_space(1))) v4i*)g,
      (__attribute__((address_space(3))) v4i*)l, 0, 0);
#else
  *(v8bf*)l = *(const v8bf*)g;
#endif
}

__device__ __forceinline__ void wait_async_le(int n) {
#if USE_ASYNC_LDS
  if (n == 4) __builtin_amdgcn_s_wait_asynccnt(4);
  else        __builtin_amdgcn_s_wait_asynccnt(0);
#endif
  (void)n;
}

// ---------------------------------------------------------------------------
// f32 -> bf16 conversion
// ---------------------------------------------------------------------------
__global__ void cvt_f32_bf16_kernel(const float* __restrict__ in,
                                    unsigned short* __restrict__ out, int n) {
  int i = blockIdx.x * blockDim.x + threadIdx.x;
  if (i < n) out[i] = f2bf(in[i]);
}

// ---------------------------------------------------------------------------
// GEMM: out[m,n] = sum_k A[m,k] * W[n,k] (+ bias[n])
// A: MxK bf16 row-major, W: NxK bf16 row-major (B^T, matching x @ W.T).
// 4 waves/block; each wave computes a 16x64 tile.
// mode 0: bf16 row-major. mode 1: bf16 transposed V layout
//   dst[((b*H+h)*D + d)*S + t], m = t*BATCH+b, n = h*HDIM+d. mode 2: f32.
// ---------------------------------------------------------------------------
__global__ __launch_bounds__(128)
void gemm_bf16_kernel(const unsigned short* __restrict__ A,
                      const unsigned short* __restrict__ W,
                      const float* __restrict__ bias,
                      void* __restrict__ out,
                      int M, int N, int K, int mode) {
  const int lane = threadIdx.x & 31;
  const int wave = threadIdx.x >> 5;
  const int half = lane >> 4;
  const int lr   = lane & 15;
  const int m0 = blockIdx.x * 16;
  const int n0 = (blockIdx.y * 4 + wave) * 64;

  v8f acc[4] = {};

  const unsigned short* arow = A + (size_t)(m0 + lr) * K;
  const unsigned short* wrow = W + (size_t)(n0 + lr) * K;

  for (int k = 0; k < K; k += 32) {
    v16bf af = load_afrag(arow + k + half * 8);
#pragma unroll
    for (int c = 0; c < 4; ++c) {
      v16bf bf = load_bfrag(wrow + (size_t)c * 16 * K + k + half * 8);
      acc[c] = wmma_bf16(af, bf, acc[c]);
    }
  }

  if (bias) {
#pragma unroll
    for (int c = 0; c < 4; ++c) {
      float bv = bias[n0 + c * 16 + lr];
#pragma unroll
      for (int r = 0; r < 8; ++r) acc[c][r] += bv;
    }
  }

  if (mode == 0) {
    unsigned short* o = (unsigned short*)out;
#pragma unroll
    for (int c = 0; c < 4; ++c)
#pragma unroll
      for (int r = 0; r < 8; ++r) {
        int row = m0 + r + half * 8;
        o[(size_t)row * N + n0 + c * 16 + lr] = f2bf(acc[c][r]);
      }
  } else if (mode == 1) {
    unsigned short* o = (unsigned short*)out;
#pragma unroll
    for (int c = 0; c < 4; ++c)
#pragma unroll
      for (int r = 0; r < 8; ++r) {
        int row = m0 + r + half * 8;       // row = t*BATCH + b
        int t = row / BATCH, b = row % BATCH;
        int col = n0 + c * 16 + lr;        // col = h*HDIM + d
        int hh = col >> 6, d = col & 63;
        o[((size_t)(b * NHEAD + hh) * HDIM + d) * S_LEN + t] = f2bf(acc[c][r]);
      }
  } else {
    float* o = (float*)out;
#pragma unroll
    for (int c = 0; c < 4; ++c)
#pragma unroll
      for (int r = 0; r < 8; ++r) {
        int row = m0 + r + half * 8;
        o[(size_t)row * N + n0 + c * 16 + lr] = acc[c][r];
      }
  }
}

// ---------------------------------------------------------------------------
// Flash attention. Block = 4 waves sharing one (b,h); each wave owns a 16-row
// query tile of a 64-row block. K/V tiles (32 keys) are double-buffered in
// LDS via async global->LDS copies (ASYNCcnt) shared by all 4 waves.
// qp/kp: [S,B,P] bf16 row-major; vT: [B,H,D,S] bf16; ao: [S,B,P] bf16.
// ---------------------------------------------------------------------------
__global__ __launch_bounds__(128)
void attn_kernel(const unsigned short* __restrict__ qp,
                 const unsigned short* __restrict__ kp,
                 const unsigned short* __restrict__ vT,
                 unsigned short* __restrict__ ao) {
  const int tid  = threadIdx.x;
  const int lane = tid & 31;
  const int wave = tid >> 5;
  const int half = lane >> 4;
  const int lr   = lane & 15;
  const int bh = blockIdx.y;      // b*NHEAD + h
  const int b  = bh >> 4;
  const int h  = bh & 15;
  const int m0 = blockIdx.x * 64 + wave * 16;
  const float scale = 0.125f;     // 1/sqrt(64)

  __shared__ unsigned short ktile[2][32 * KSTRIDE];  // 32 keys x 64 dims
  __shared__ unsigned short vtile[2][64 * VSTRIDE];  // 64 dims x 32 keys
  __shared__ unsigned short pstage[4][16 * PSTRIDE]; // per-wave P transpose
  unsigned short* pbuf = pstage[wave];

  const unsigned short* kbase = kp + (size_t)b * PDIM + h * HDIM;
  const unsigned short* vbase = vT + (size_t)bh * HDIM * S_LEN;

  // cooperative prefetch of one 32-key K/V tile (4 async b128 ops per thread)
  auto prefetch = [&](int tt, int buf) {
#pragma unroll
    for (int j = 0; j < 2; ++j) {
      int c = tid + j * 128;                 // 256 chunks of 16B each
      {                                      // K: 32 rows x 64 elems
        int row = c >> 3, o = (c & 7) * 8;
        copy16_g2l(kbase + (size_t)(tt + row) * (BATCH * PDIM) + o,
                   &ktile[buf][row * KSTRIDE + o]);
      }
      {                                      // V: 64 rows x 32 elems
        int d = c >> 2, o = (c & 3) * 8;
        copy16_g2l(vbase + (size_t)d * S_LEN + tt + o,
                   &vtile[buf][d * VSTRIDE + o]);
      }
    }
  };

  // Q fragments: K = 64 -> 2 chunks of 32
  const unsigned short* qrow =
      qp + ((size_t)(m0 + lr) * BATCH + b) * PDIM + h * HDIM;
  v16bf qf0 = load_afrag(qrow + 0  + half * 8);
  v16bf qf1 = load_afrag(qrow + 32 + half * 8);

  v8f O[4] = {};
  float mrun[8], lrun[8];
#pragma unroll
  for (int r = 0; r < 8; ++r) { mrun[r] = -3.0e38f; lrun[r] = 0.0f; }

  prefetch(0, 0);

  for (int i = 0; i < S_LEN / 32; ++i) {
    const int tt = i * 32;
    if (tt + 32 < S_LEN) {
      prefetch(tt + 32, (i + 1) & 1);
      wait_async_le(4);                 // tile i done; tile i+1 in flight
    } else {
      wait_async_le(0);
    }
    __syncthreads();                    // K/V tile i visible to all waves

    const unsigned short* kt = ktile[i & 1];
    const unsigned short* vt = vtile[i & 1];

    // scores: Q(16x64) x K^T -> two 16x16 tiles, from LDS
    v8f s0 = {}, s1 = {};
    {
      const unsigned short* k0 = kt + (size_t)lr * KSTRIDE;
      const unsigned short* k1 = kt + (size_t)(16 + lr) * KSTRIDE;
      v16bf b00 = load_bfrag(k0 + 0  + half * 8);
      v16bf b01 = load_bfrag(k0 + 32 + half * 8);
      v16bf b10 = load_bfrag(k1 + 0  + half * 8);
      v16bf b11 = load_bfrag(k1 + 32 + half * 8);
      s0 = wmma_bf16(qf0, b00, s0);
      s0 = wmma_bf16(qf1, b01, s0);
      s1 = wmma_bf16(qf0, b10, s1);
      s1 = wmma_bf16(qf1, b11, s1);
    }

    // online softmax. C-layout: VGPR r -> row r (lo half) / r+8 (hi half);
    // columns striped over the 16 lanes of each half.
    float alpha[8];
#pragma unroll
    for (int r = 0; r < 8; ++r) {
      s0[r] *= scale;
      s1[r] *= scale;
      float rm = hmax16(fmaxf(s0[r], s1[r]));
      float mn = fmaxf(mrun[r], rm);
      alpha[r] = __expf(mrun[r] - mn);
      mrun[r] = mn;
      s0[r] = __expf(s0[r] - mn);
      s1[r] = __expf(s1[r] - mn);
      float rs = hsum16(s0[r] + s1[r]);
      lrun[r] = lrun[r] * alpha[r] + rs;
    }
#pragma unroll
    for (int c = 0; c < 4; ++c)
#pragma unroll
      for (int r = 0; r < 8; ++r) O[c][r] *= alpha[r];

    // stage P (bf16) through per-wave LDS: C-layout -> A-fragment layout.
    // Same-wave DS ops execute in order, so no barrier needed here.
#pragma unroll
    for (int r = 0; r < 8; ++r) {
      int row = r + half * 8;
      pbuf[row * PSTRIDE + lr]      = f2bf(s0[r]);
      pbuf[row * PSTRIDE + 16 + lr] = f2bf(s1[r]);
    }
    v16bf pa = load_afrag(pbuf + lr * PSTRIDE + half * 8);

    // O += P(16x32) x V(32x64), V fragments from LDS
#pragma unroll
    for (int c = 0; c < 4; ++c) {
      v16bf vb = load_bfrag(vt + (size_t)(c * 16 + lr) * VSTRIDE + half * 8);
      O[c] = wmma_bf16(pa, vb, O[c]);
    }
    __syncthreads();                    // all waves done with tile i buffers
  }

  // epilogue: normalize by l, write ao[(s*B+b)*P + h*64 + d] as bf16
  float inv[8];
#pragma unroll
  for (int r = 0; r < 8; ++r) inv[r] = 1.0f / lrun[r];
#pragma unroll
  for (int c = 0; c < 4; ++c)
#pragma unroll
    for (int r = 0; r < 8; ++r) {
      int row = m0 + r + half * 8;
      ao[((size_t)row * BATCH + b) * PDIM + h * HDIM + c * 16 + lr] =
          f2bf(O[c][r] * inv[r]);
    }
}

// ---------------------------------------------------------------------------
// launch
// ---------------------------------------------------------------------------
extern "C" void kernel_launch(void* const* d_in, const int* in_sizes, int n_in,
                              void* d_out, int out_size, void* d_ws, size_t ws_size,
                              hipStream_t stream) {
  const float* q  = (const float*)d_in[0];
  const float* v  = (const float*)d_in[1];
  const float* k  = (const float*)d_in[2];
  const float* Wq = (const float*)d_in[3];
  const float* bq = (const float*)d_in[4];
  const float* Wk = (const float*)d_in[5];
  const float* bk = (const float*)d_in[6];
  const float* Wv = (const float*)d_in[7];
  const float* bv = (const float*)d_in[8];
  const float* Wo = (const float*)d_in[9];

  const size_t SBF = (size_t)MROWS * FDIM;   // 8,388,608 elems
  const size_t WSZ = (size_t)PDIM * FDIM;    // 1,048,576 elems

  char* ws = (char*)d_ws;
  size_t off = 0;
  unsigned short* qbf = (unsigned short*)(ws + off); off += SBF * 2;
  unsigned short* vbf = (unsigned short*)(ws + off); off += SBF * 2;
  unsigned short* kbf = (unsigned short*)(ws + off); off += SBF * 2;
  unsigned short* wq  = (unsigned short*)(ws + off); off += WSZ * 2;
  unsigned short* wk  = (unsigned short*)(ws + off); off += WSZ * 2;
  unsigned short* wv  = (unsigned short*)(ws + off); off += WSZ * 2;
  unsigned short* wo  = (unsigned short*)(ws + off); off += WSZ * 2;
  unsigned short* qp  = (unsigned short*)(ws + off); off += SBF * 2;
  unsigned short* kp  = (unsigned short*)(ws + off); off += SBF * 2;
  unsigned short* vT  = (unsigned short*)(ws + off); off += SBF * 2;
  unsigned short* aob = (unsigned short*)(ws + off); off += SBF * 2;

  // 1) f32 -> bf16 conversions
  {
    int n = (int)SBF;
    cvt_f32_bf16_kernel<<<n / 256, 256, 0, stream>>>(q, qbf, n);
    cvt_f32_bf16_kernel<<<n / 256, 256, 0, stream>>>(v, vbf, n);
    cvt_f32_bf16_kernel<<<n / 256, 256, 0, stream>>>(k, kbf, n);
    int w = (int)WSZ;
    cvt_f32_bf16_kernel<<<w / 256, 256, 0, stream>>>(Wq, wq, w);
    cvt_f32_bf16_kernel<<<w / 256, 256, 0, stream>>>(Wk, wk, w);
    cvt_f32_bf16_kernel<<<w / 256, 256, 0, stream>>>(Wv, wv, w);
    cvt_f32_bf16_kernel<<<w / 256, 256, 0, stream>>>(Wo, wo, w);
  }

  // 2) projections
  dim3 ggrid(MROWS / 16, PDIM / 256);
  gemm_bf16_kernel<<<ggrid, 128, 0, stream>>>(qbf, wq, bq, qp,
                                              MROWS, PDIM, FDIM, 0);
  gemm_bf16_kernel<<<ggrid, 128, 0, stream>>>(kbf, wk, bk, kp,
                                              MROWS, PDIM, FDIM, 0);
  gemm_bf16_kernel<<<ggrid, 128, 0, stream>>>(vbf, wv, bv, vT,
                                              MROWS, PDIM, FDIM, 1);

  // 3) attention
  dim3 agrid(S_LEN / 64, BATCH * NHEAD);
  attn_kernel<<<agrid, 128, 0, stream>>>(qp, kp, vT, aob);

  // 4) output projection (f32 out, no bias)
  dim3 ogrid(MROWS / 16, FDIM / 256);
  gemm_bf16_kernel<<<ogrid, 128, 0, stream>>>(aob, wo, nullptr, d_out,
                                              MROWS, FDIM, PDIM, 2);
}
